// StateFusionMamba_13606456394023
// MI455X (gfx1250) — matrix operand order repaired and checked by reading, not compiled, not run on hardware
//
#include <hip/hip_runtime.h>
#include <hip/hip_bf16.h>

// ---------------------------------------------------------------------------
// StateFusionMamba for MI455X (gfx1250, wave32, WMMA).
// Dense projections: bf16 WMMA, 128x128 block tile, 32x64 wave tile
// (2x4 accumulators -> 8 v_wmma per wave per K-step), double-buffered LDS
// filled with CDNA5 async global->LDS copies (ASYNCcnt).
// Selective scan: one lane per (batch, channel), 16 states in registers.
// ---------------------------------------------------------------------------

#define B_SZ     16
#define K_SEQ    512
#define D_IN     512
#define D_MODEL  1024
#define D_OUT    512
#define N_LAYERS 2
#define D_INNER  2048
#define D_STATE  16
#define D_CONV   4
#define DT_RANK  64
#define L_SEQ    (K_SEQ + 1)            // 513
#define ROWS     (B_SZ * L_SEQ)          // 8208 (multiple of 16)
#define EPS_LN   1e-5f

typedef __attribute__((ext_vector_type(16))) __bf16 v16bf;
typedef __attribute__((ext_vector_type(8)))  float  v8f;
typedef __attribute__((ext_vector_type(4)))  int    v4i;

union V16 { uint4 q[2]; v16bf v; };

__device__ __forceinline__ __bf16 f2bf(float f) { return (__bf16)f; }
__device__ __forceinline__ float silu(float x) { return x / (1.0f + __expf(-x)); }

// ---- CDNA5 async global->LDS copy, probed (safe fallback) -----------------
#if __has_builtin(__builtin_amdgcn_global_load_async_to_lds_b128)
#define HAVE_ASYNC_LDS 1
#else
#define HAVE_ASYNC_LDS 0
#endif

__device__ __forceinline__ void copy16_g2l(const uint4* gsrc, uint4* ldst) {
#if HAVE_ASYNC_LDS
    __builtin_amdgcn_global_load_async_to_lds_b128((v4i*)gsrc, (v4i*)ldst, 0, 0);
#else
    *ldst = *gsrc;
#endif
}

__device__ __forceinline__ void wait_async_copies() {
#if HAVE_ASYNC_LDS
#if __has_builtin(__builtin_amdgcn_s_wait_asynccnt)
    __builtin_amdgcn_s_wait_asynccnt(0);
#else
    asm volatile("s_wait_asynccnt 0" ::: "memory");
#endif
#endif
}

// ---------------------------------------------------------------------------
// Generic f32 -> bf16 conversion (grid-stride).
// ---------------------------------------------------------------------------
__global__ __launch_bounds__(256)
void cvt_f32_bf16_kernel(const float* __restrict__ in, __bf16* __restrict__ out,
                         size_t n) {
    size_t i = (size_t)blockIdx.x * blockDim.x + threadIdx.x;
    size_t stride = (size_t)gridDim.x * blockDim.x;
    for (; i < n; i += stride) out[i] = f2bf(in[i]);
}

// ---------------------------------------------------------------------------
// Mask padded states by length; emit bf16 (B,K,D_IN).
// ---------------------------------------------------------------------------
__global__ __launch_bounds__(256)
void pack_states_kernel(const float* __restrict__ states,
                        const int* __restrict__ lengths,
                        __bf16* __restrict__ out, size_t n) {
    size_t i = (size_t)blockIdx.x * blockDim.x + threadIdx.x;
    if (i >= n) return;
    int k = (int)((i / D_IN) % K_SEQ);
    int b = (int)(i / ((size_t)D_IN * K_SEQ));
    float v = (k < lengths[b]) ? states[i] : 0.0f;
    out[i] = f2bf(v);
}

// ---------------------------------------------------------------------------
// WMMA GEMM:  C[M,N] = act( A[M,K](bf16,lda) * W[N,K]^T(bf16,ldw) + bias )
//
// block: 256 thr = 8 waves in a 4(M) x 2(N) grid. Block tile 128x128.
// Wave tile 32x64 = 2x4 WMMA tiles -> 8 v_wmma_f32_16x16x32_bf16 per K-step
// with register-level fragment reuse. LDS double-buffered; rows padded to 40
// halfs (80B) so the 16 fragment rows land in distinct bank groups.
// Requires: K % 32 == 0, A/W rows 16B-aligned (true for all call sites).
// act: 0=none, 1=silu, 2=softplus
// ---------------------------------------------------------------------------
#define LDSPITCH 40

__global__ __launch_bounds__(256)
void wmma_gemm_bf16_kernel(const __bf16* __restrict__ A, int lda,
                           const __bf16* __restrict__ W, int ldw,
                           const float* __restrict__ bias,
                           float* __restrict__ C, int ldc,
                           int M, int N, int Ksz, int act) {
    __shared__ __bf16 Ash[2][128][LDSPITCH];   // ~20 KB
    __shared__ __bf16 Bsh[2][128][LDSPITCH];   // ~20 KB

    const int tid  = threadIdx.x;
    const int wave = tid >> 5;
    const int lane = tid & 31;
    const int wm   = wave & 3;        // 0..3 : 32-row strip within block
    const int wn   = wave >> 2;       // 0..1 : 64-col strip within block
    const int m0   = blockIdx.y * 128;
    const int nblk = blockIdx.x * 128;

    v8f acc[2][4];
#pragma unroll
    for (int i = 0; i < 2; ++i)
#pragma unroll
        for (int j = 0; j < 4; ++j) acc[i][j] = (v8f){};

    const int nk = Ksz >> 5;

    auto stage = [&](int ks, int buf) {
        const int k0 = ks << 5;
        // A tile: 128 rows x 32 halfs = 512 x uint4; 2 per thread
#pragma unroll
        for (int e = tid; e < 512; e += 256) {
            int r = e >> 2, qc = e & 3;
            int row = m0 + r; if (row >= M) row = M - 1;
            copy16_g2l((const uint4*)(A + (size_t)row * lda + k0 + qc * 8),
                       (uint4*)&Ash[buf][r][qc * 8]);
        }
        // W tile: 128 cols x 32 halfs
#pragma unroll
        for (int e = tid; e < 512; e += 256) {
            int cn = e >> 2, qc = e & 3;
            int col = nblk + cn; if (col >= N) col = N - 1;
            copy16_g2l((const uint4*)(W + (size_t)col * ldw + k0 + qc * 8),
                       (uint4*)&Bsh[buf][cn][qc * 8]);
        }
    };

    auto compute = [&](int buf) {
        // A fragment (16x32): lanes 0-15 -> M=lane, K {0..7,16..23};
        //                     lanes 16-31 -> M=lane-16, K {8..15,24..31}
        const int mrow = lane & 15;
        const int kh   = (lane >> 4) * 8;
        V16 af[2];
#pragma unroll
        for (int i = 0; i < 2; ++i) {
            int r = wm * 32 + i * 16 + mrow;
            af[i].q[0] = *(const uint4*)&Ash[buf][r][kh];
            af[i].q[1] = *(const uint4*)&Ash[buf][r][16 + kh];
        }
        // B fragment (32x16): lanes 0-15 -> col=lane, K {0..7,16..23};
        //                     lanes 16-31 -> col=lane-16, K {8..15,24..31}
        const int kb = (lane >> 4) * 8;
        V16 bf4[4];
#pragma unroll
        for (int j = 0; j < 4; ++j) {
            int c = wn * 64 + j * 16 + (lane & 15);
            bf4[j].q[0] = *(const uint4*)&Bsh[buf][c][kb];
            bf4[j].q[1] = *(const uint4*)&Bsh[buf][c][kb + 16];
        }
#pragma unroll
        for (int i = 0; i < 2; ++i)
#pragma unroll
            for (int j = 0; j < 4; ++j)
                acc[i][j] = __builtin_amdgcn_wmma_f32_16x16x32_bf16(
                    false, af[i].v, false, bf4[j].v, (short)0, acc[i][j],
                    false, false);
    };

    stage(0, 0);
    wait_async_copies();
    __syncthreads();
    for (int kk = 0; kk < nk; ++kk) {
        const int cur = kk & 1;
        if (kk + 1 < nk) stage(kk + 1, cur ^ 1);
        compute(cur);
        wait_async_copies();
        __syncthreads();
    }

    // ---- epilogue: C/D tile layout: VGPR r -> row r + 8*(lane>=16), col=lane&15
    const int ncl   = lane & 15;
    const int mbase = (lane >> 4) * 8;
#pragma unroll
    for (int j = 0; j < 4; ++j) {
        int col = nblk + wn * 64 + j * 16 + ncl;
        if (col >= N) continue;
        float bv = bias ? bias[col] : 0.0f;
#pragma unroll
        for (int i = 0; i < 2; ++i) {
            int rb = m0 + wm * 32 + i * 16 + mbase;
#pragma unroll
            for (int r = 0; r < 8; ++r) {
                int row = rb + r;
                if (row < M) {
                    float v = acc[i][j][r] + bv;
                    if (act == 1)      v = silu(v);
                    else if (act == 2) v = (v > 20.0f) ? v : log1pf(__expf(v));
                    C[(size_t)row * ldc + col] = v;
                }
            }
        }
    }
}

// ---------------------------------------------------------------------------
// Assemble X: X[b,0,:] = cls; X[b,1+k,:] = T0[b*K+k,:]; plus bf16 copy.
// ---------------------------------------------------------------------------
__global__ __launch_bounds__(256)
void build_x_kernel(const float* __restrict__ T0, const float* __restrict__ cls,
                    float* __restrict__ X, __bf16* __restrict__ Xbf, size_t n) {
    size_t i = (size_t)blockIdx.x * blockDim.x + threadIdx.x;
    if (i >= n) return;
    int c   = (int)(i % D_MODEL);
    int row = (int)(i / D_MODEL);
    int b = row / L_SEQ, l = row % L_SEQ;
    float v = (l == 0) ? cls[c] : T0[((size_t)b * K_SEQ + (l - 1)) * D_MODEL + c];
    X[i] = v;
    Xbf[i] = f2bf(v);
}

__global__ __launch_bounds__(256)
void x_to_bf16_kernel(const float* __restrict__ X, __bf16* __restrict__ Xbf,
                      size_t n) {
    size_t i = (size_t)blockIdx.x * blockDim.x + threadIdx.x;
    size_t stride = (size_t)gridDim.x * blockDim.x;
    for (; i < n; i += stride) Xbf[i] = f2bf(X[i]);
}

// ---------------------------------------------------------------------------
// Causal depthwise conv (width 4) + bias + SiLU over xi = XZ[:, :D_INNER].
// ---------------------------------------------------------------------------
__global__ __launch_bounds__(256)
void conv_silu_kernel(const float* __restrict__ XZ,
                      const float* __restrict__ cw,   // [D_INNER, 4]
                      const float* __restrict__ cb,   // [D_INNER]
                      float* __restrict__ XC, __bf16* __restrict__ XCbf,
                      size_t n) {
    size_t i = (size_t)blockIdx.x * blockDim.x + threadIdx.x;
    if (i >= n) return;
    int d   = (int)(i % D_INNER);
    int row = (int)(i / D_INNER);
    int b = row / L_SEQ, l = row % L_SEQ;
    float s = cb[d];
#pragma unroll
    for (int j = 0; j < D_CONV; ++j) {
        int ll = l - (D_CONV - 1) + j;
        if (ll >= 0)
            s += cw[d * D_CONV + j] *
                 XZ[((size_t)b * L_SEQ + ll) * (2 * D_INNER) + d];
    }
    float v = silu(s);
    XC[i] = v;
    XCbf[i] = f2bf(v);
}

// ---------------------------------------------------------------------------
// Selective scan: one lane per (b, d); fused D-residual and SiLU(z) gate.
// ---------------------------------------------------------------------------
__global__ __launch_bounds__(256)
void selective_scan_kernel(const float* __restrict__ DT,
                           const float* __restrict__ XDBL,   // [rows, 96]
                           const float* __restrict__ XC,
                           const float* __restrict__ XZ,
                           const float* __restrict__ Alog,   // [D_INNER, 16]
                           const float* __restrict__ Dp,     // [D_INNER]
                           float* __restrict__ Y, __bf16* __restrict__ Ybf) {
    int idx = blockIdx.x * blockDim.x + threadIdx.x;   // 0 .. B*D_INNER-1
    if (idx >= B_SZ * D_INNER) return;
    int b = idx >> 11;
    int d = idx & (D_INNER - 1);

    float Arow[D_STATE];
#pragma unroll
    for (int n = 0; n < D_STATE; ++n) Arow[n] = -__expf(Alog[d * D_STATE + n]);
    float Dpar = Dp[d];
    float h[D_STATE];
#pragma unroll
    for (int n = 0; n < D_STATE; ++n) h[n] = 0.0f;

    for (int l = 0; l < L_SEQ; ++l) {
        size_t r  = (size_t)b * L_SEQ + l;
        float dt  = DT[r * D_INNER + d];
        float x   = XC[r * D_INNER + d];
        float z   = XZ[r * (2 * D_INNER) + D_INNER + d];
        const float* bc = &XDBL[r * (DT_RANK + 2 * D_STATE) + DT_RANK];
        float dtx = dt * x;
        float y = 0.0f;
#pragma unroll
        for (int n = 0; n < D_STATE; ++n) {
            float dA = __expf(dt * Arow[n]);
            h[n] = dA * h[n] + dtx * bc[n];
            y += h[n] * bc[D_STATE + n];
        }
        y = (y + x * Dpar) * silu(z);
        Y[r * D_INNER + d]   = y;
        Ybf[r * D_INNER + d] = f2bf(y);
    }
}

// ---------------------------------------------------------------------------
// LayerNorm of the cls row only (the head consumes token 0 only); bf16 out.
// ---------------------------------------------------------------------------
__global__ __launch_bounds__(256)
void layernorm_cls_kernel(const float* __restrict__ X,
                          const float* __restrict__ g,
                          const float* __restrict__ be,
                          __bf16* __restrict__ featbf) {
    __shared__ float red[256];
    int b = blockIdx.x, tid = threadIdx.x;
    const float* row = X + (size_t)b * L_SEQ * D_MODEL;

    float s = 0.0f;
    for (int c = tid; c < D_MODEL; c += 256) s += row[c];
    red[tid] = s; __syncthreads();
    for (int k = 128; k > 0; k >>= 1) {
        if (tid < k) red[tid] += red[tid + k];
        __syncthreads();
    }
    float mu = red[0] / (float)D_MODEL;
    __syncthreads();

    float v = 0.0f;
    for (int c = tid; c < D_MODEL; c += 256) {
        float t = row[c] - mu; v += t * t;
    }
    red[tid] = v; __syncthreads();
    for (int k = 128; k > 0; k >>= 1) {
        if (tid < k) red[tid] += red[tid + k];
        __syncthreads();
    }
    float rstd = rsqrtf(red[0] / (float)D_MODEL + EPS_LN);

    for (int c = tid; c < D_MODEL; c += 256)
        featbf[b * D_MODEL + c] = f2bf((row[c] - mu) * rstd * g[c] + be[c]);
}

// ---------------------------------------------------------------------------
// Host-side orchestration
// ---------------------------------------------------------------------------
static inline dim3 gemm_grid(int M, int N) {
    return dim3((N + 127) / 128, (M + 127) / 128);
}

extern "C" void kernel_launch(void* const* d_in, const int* in_sizes, int n_in,
                              void* d_out, int out_size, void* d_ws, size_t ws_size,
                              hipStream_t stream) {
    (void)in_sizes; (void)n_in; (void)out_size; (void)ws_size;

    const float* states   = (const float*)d_in[0];
    const int*   lengths  = (const int*)  d_in[1];
    const float* mlp_w    = (const float*)d_in[2];
    const float* mlp_b    = (const float*)d_in[3];
    const float* cls_tok  = (const float*)d_in[4];
    const float* in_projw = (const float*)d_in[5];
    const float* conv_w   = (const float*)d_in[6];
    const float* conv_b   = (const float*)d_in[7];
    const float* x_projw  = (const float*)d_in[8];
    const float* dt_projw = (const float*)d_in[9];
    const float* dt_projb = (const float*)d_in[10];
    const float* A_log    = (const float*)d_in[11];
    const float* D_param  = (const float*)d_in[12];
    const float* out_prw  = (const float*)d_in[13];
    const float* norm_g   = (const float*)d_in[14];
    const float* norm_b   = (const float*)d_in[15];
    const float* proj_w   = (const float*)d_in[16];
    const float* proj_b   = (const float*)d_in[17];

    size_t off = 0;
    auto alloc = [&](size_t bytes) -> void* {
        void* p = (char*)d_ws + off;
        off = (off + bytes + 255) & ~(size_t)255;
        return p;
    };
    __bf16* mlpWb  = (__bf16*)alloc((size_t)D_MODEL * D_IN * 2);
    __bf16* inWb   = (__bf16*)alloc((size_t)N_LAYERS * 2 * D_INNER * D_MODEL * 2);
    __bf16* xpWb   = (__bf16*)alloc((size_t)N_LAYERS * (DT_RANK + 2 * D_STATE) * D_INNER * 2);
    __bf16* dtWb   = (__bf16*)alloc((size_t)N_LAYERS * D_INNER * DT_RANK * 2);
    __bf16* outWb  = (__bf16*)alloc((size_t)N_LAYERS * D_MODEL * D_INNER * 2);
    __bf16* pWb    = (__bf16*)alloc((size_t)D_OUT * D_MODEL * 2);
    __bf16* Sbf    = (__bf16*)alloc((size_t)B_SZ * K_SEQ * D_IN * 2);
    float*  T0     = (float*) alloc((size_t)B_SZ * K_SEQ * D_MODEL * 4);
    float*  X      = (float*) alloc((size_t)ROWS * D_MODEL * 4);
    __bf16* Xbf    = (__bf16*)alloc((size_t)ROWS * D_MODEL * 2);
    float*  XZ     = (float*) alloc((size_t)ROWS * 2 * D_INNER * 4);
    float*  XC     = (float*) alloc((size_t)ROWS * D_INNER * 4);
    __bf16* XCbf   = (__bf16*)alloc((size_t)ROWS * D_INNER * 2);
    float*  XDBL   = (float*) alloc((size_t)ROWS * (DT_RANK + 2 * D_STATE) * 4);
    __bf16* XDBLb  = (__bf16*)alloc((size_t)ROWS * (DT_RANK + 2 * D_STATE) * 2);
    float*  DTbuf  = (float*) alloc((size_t)ROWS * D_INNER * 4);
    float*  Y      = (float*) alloc((size_t)ROWS * D_INNER * 4);
    __bf16* Ybf    = (__bf16*)alloc((size_t)ROWS * D_INNER * 2);
    __bf16* featbf = (__bf16*)alloc((size_t)B_SZ * D_MODEL * 2);

    auto cvt = [&](const float* src, __bf16* dst, size_t n) {
        int blocks = (int)((n + 255) / 256); if (blocks > 16384) blocks = 16384;
        cvt_f32_bf16_kernel<<<blocks, 256, 0, stream>>>(src, dst, n);
    };

    // 1) weights -> bf16
    cvt(mlp_w,    mlpWb, (size_t)D_MODEL * D_IN);
    cvt(in_projw, inWb,  (size_t)N_LAYERS * 2 * D_INNER * D_MODEL);
    cvt(x_projw,  xpWb,  (size_t)N_LAYERS * (DT_RANK + 2 * D_STATE) * D_INNER);
    cvt(dt_projw, dtWb,  (size_t)N_LAYERS * D_INNER * DT_RANK);
    cvt(out_prw,  outWb, (size_t)N_LAYERS * D_MODEL * D_INNER);
    cvt(proj_w,   pWb,   (size_t)D_OUT * D_MODEL);

    // 2) mask states, MLP embed, prepend cls
    {
        size_t n = (size_t)B_SZ * K_SEQ * D_IN;
        pack_states_kernel<<<(int)((n + 255) / 256), 256, 0, stream>>>(
            states, lengths, Sbf, n);
        wmma_gemm_bf16_kernel<<<gemm_grid(B_SZ * K_SEQ, D_MODEL), 256, 0, stream>>>(
            Sbf, D_IN, mlpWb, D_IN, mlp_b, T0, D_MODEL,
            B_SZ * K_SEQ, D_MODEL, D_IN, 0);
        size_t nx = (size_t)ROWS * D_MODEL;
        build_x_kernel<<<(int)((nx + 255) / 256), 256, 0, stream>>>(
            T0, cls_tok, X, Xbf, nx);
    }

    // 3) Mamba layers
    for (int i = 0; i < N_LAYERS; ++i) {
        const __bf16* inW  = inWb  + (size_t)i * 2 * D_INNER * D_MODEL;
        const __bf16* xpW  = xpWb  + (size_t)i * (DT_RANK + 2 * D_STATE) * D_INNER;
        const __bf16* dtW  = dtWb  + (size_t)i * D_INNER * DT_RANK;
        const __bf16* outW = outWb + (size_t)i * D_MODEL * D_INNER;
        const float*  cw   = conv_w   + (size_t)i * D_INNER * D_CONV;
        const float*  cb   = conv_b   + (size_t)i * D_INNER;
        const float*  dpb  = dt_projb + (size_t)i * D_INNER;
        const float*  Al   = A_log    + (size_t)i * D_INNER * D_STATE;
        const float*  Dpr  = D_param  + (size_t)i * D_INNER;

        // in_proj: XZ = X @ inW^T
        wmma_gemm_bf16_kernel<<<gemm_grid(ROWS, 2 * D_INNER), 256, 0, stream>>>(
            Xbf, D_MODEL, inW, D_MODEL, nullptr, XZ, 2 * D_INNER,
            ROWS, 2 * D_INNER, D_MODEL, 0);

        // causal conv + SiLU
        size_t nc = (size_t)ROWS * D_INNER;
        conv_silu_kernel<<<(int)((nc + 255) / 256), 256, 0, stream>>>(
            XZ, cw, cb, XC, XCbf, nc);

        // x_proj: XDBL = XC @ xpW^T   (dt | B | C)
        wmma_gemm_bf16_kernel<<<gemm_grid(ROWS, DT_RANK + 2 * D_STATE), 256, 0, stream>>>(
            XCbf, D_INNER, xpW, D_INNER, nullptr, XDBL, DT_RANK + 2 * D_STATE,
            ROWS, DT_RANK + 2 * D_STATE, D_INNER, 0);
        cvt(XDBL, XDBLb, (size_t)ROWS * (DT_RANK + 2 * D_STATE));

        // dt_proj + softplus (A strided: lda = 96, K = 64)
        wmma_gemm_bf16_kernel<<<gemm_grid(ROWS, D_INNER), 256, 0, stream>>>(
            XDBLb, DT_RANK + 2 * D_STATE, dtW, DT_RANK, dpb, DTbuf, D_INNER,
            ROWS, D_INNER, DT_RANK, 2);

        // selective scan + D residual + SiLU(z) gate
        selective_scan_kernel<<<(B_SZ * D_INNER) / 256, 256, 0, stream>>>(
            DTbuf, XDBL, XC, XZ, Al, Dpr, Y, Ybf);

        // out_proj: X = Y @ outW^T ; refresh bf16 X
        wmma_gemm_bf16_kernel<<<gemm_grid(ROWS, D_MODEL), 256, 0, stream>>>(
            Ybf, D_INNER, outW, D_INNER, nullptr, X, D_MODEL,
            ROWS, D_MODEL, D_INNER, 0);
        size_t nx = (size_t)ROWS * D_MODEL;
        x_to_bf16_kernel<<<8192, 256, 0, stream>>>(X, Xbf, nx);
    }

    // 4) LayerNorm on cls rows, final projection -> d_out
    layernorm_cls_kernel<<<B_SZ, 256, 0, stream>>>(X, norm_g, norm_b, featbf);
    wmma_gemm_bf16_kernel<<<gemm_grid(B_SZ, D_OUT), 256, 0, stream>>>(
        featbf, D_MODEL, pWb, D_MODEL, proj_b, (float*)d_out, D_OUT,
        B_SZ, D_OUT, D_MODEL, 0);
}
